// GraphConvModel_13597866459243
// MI455X (gfx1250) — matrix operand order repaired and checked by reading, not compiled, not run on hardware
//
#include <hip/hip_runtime.h>
#include <math.h>

typedef __attribute__((ext_vector_type(16))) __bf16 v16bf;
typedef __attribute__((ext_vector_type(8)))  float  v8f;

#define DF 128   // feature / embedding dim (F == D == 128)

// ---------------------------------------------------------------- zero fill
__global__ void zero_f32(float* __restrict__ p, long n) {
  long i = ((long)blockIdx.x * blockDim.x + threadIdx.x) * 4;
  if (i + 3 < n) {
    *(float4*)(p + i) = make_float4(0.f, 0.f, 0.f, 0.f);
  } else {
    for (long j = i; j < n; ++j) p[j] = 0.f;
  }
}

// ------------------------------------------------- edge scatter-add (segment_sum)
// thread t: edge = t>>5, features f..f+3 ; agg[dst] += h[src]
__global__ void edge_scatter(const float* __restrict__ h,
                             const int* __restrict__ src,
                             const int* __restrict__ dst,
                             float* __restrict__ agg, int E) {
  int t = blockIdx.x * blockDim.x + threadIdx.x;
  int e = t >> 5;
  if (e >= E) return;
  int f = (t & 31) << 2;
  int s = src[e], d = dst[e];
  const float4 v = *(const float4*)(h + (size_t)s * DF + f);
  float* p = agg + (size_t)d * DF + f;
  atomicAdd(p + 0, v.x);
  atomicAdd(p + 1, v.y);
  atomicAdd(p + 2, v.z);
  atomicAdd(p + 3, v.w);
}

// ---------------------------------------------------------- f32 -> bf16 (RNE)
__device__ __forceinline__ unsigned int pk_bf16(float a, float b) {
  unsigned int ua = __float_as_uint(a);
  ua += 0x7FFFu + ((ua >> 16) & 1u);
  unsigned int ub = __float_as_uint(b);
  ub += 0x7FFFu + ((ub >> 16) & 1u);
  return (ua >> 16) | (ub & 0xFFFF0000u);
}

__global__ void cvt_f32_bf16(const float* __restrict__ in,
                             unsigned int* __restrict__ out, long n4) {
  long t = (long)blockIdx.x * blockDim.x + threadIdx.x;
  if (t >= n4) return;
  float4 v = ((const float4*)in)[t];
  uint2 o;
  o.x = pk_bf16(v.x, v.y);
  o.y = pk_bf16(v.z, v.w);
  ((uint2*)out)[t] = o;
}

// --------------------------------------------------------------- WMMA GEMM
// hout[m,n] = gelu( sum_k hb[m,k]*Wroot[n,k] + sum_k aggb[m,k]*Wrel[n,k] + bias[n] )
// One wave32 -> one 16x16 D tile via v_wmma_f32_16x16x32_bf16 (K=128 -> 4 steps x 2 GEMMs).
// A layout (16x32 bf16): lane = hi*16+hl, row m0+hl; elems 0..7 <- k=kb+hi*8..,
// elems 8..15 <- k=kb+16+hi*8..   (ISA 7.12.2)
// B layout (32x16): from W row-major [n,k]; lane col n0+hl; same k order as A
__global__ void __launch_bounds__(256)
wmma_graphconv(const unsigned short* __restrict__ hb,
               const unsigned short* __restrict__ aggb,
               const unsigned short* __restrict__ wb,   // [0,128,128]=Wroot, [1,..]=Wrel (bf16)
               const float* __restrict__ bias,
               float* __restrict__ hout, int nrows) {
  const int lane = threadIdx.x & 31;
  const int wave = threadIdx.x >> 5;   // 0..7 -> column tile
  const int hl   = lane & 15;
  const int hi   = lane >> 4;
  const int m0   = blockIdx.x << 4;
  const int n0   = wave << 4;
  if (m0 >= nrows) return;

  const unsigned short* wroot = wb;
  const unsigned short* wrel  = wb + DF * DF;

  union U { v16bf v; uint4 q[2]; };
  v8f c = {};
  const size_t arow = (size_t)(m0 + hl) * DF;
  const size_t brow = (size_t)(n0 + hl) * DF;

#pragma unroll
  for (int kb = 0; kb < DF; kb += 32) {
    U a, b;
    { const unsigned short* p = hb + arow + kb + hi * 8;
      a.q[0] = *(const uint4*)(p);  a.q[1] = *(const uint4*)(p + 16); }
    { const unsigned short* p = wroot + brow + kb + hi * 8;
      b.q[0] = *(const uint4*)(p);  b.q[1] = *(const uint4*)(p + 16); }
    c = __builtin_amdgcn_wmma_f32_16x16x32_bf16(false, a.v, false, b.v,
                                                (short)0, c, false, false);
    { const unsigned short* p = aggb + arow + kb + hi * 8;
      a.q[0] = *(const uint4*)(p);  a.q[1] = *(const uint4*)(p + 16); }
    { const unsigned short* p = wrel + brow + kb + hi * 8;
      b.q[0] = *(const uint4*)(p);  b.q[1] = *(const uint4*)(p + 16); }
    c = __builtin_amdgcn_wmma_f32_16x16x32_bf16(false, a.v, false, b.v,
                                                (short)0, c, false, false);
  }

  const float bn = bias[n0 + hl];
#pragma unroll
  for (int r = 0; r < 8; ++r) {
    float xv = c[r] + bn;                 // C/D layout: VGPR r -> row r+hi*8
    float g = 0.5f * xv * (1.0f + erff(xv * 0.70710678118654752440f));
    hout[(size_t)(m0 + r + hi * 8) * DF + n0 + hl] = g;
  }
}

// ------------------------------------------------------------- mean pool (scatter)
__global__ void pool_scatter(const float* __restrict__ h,
                             const int* __restrict__ batch,
                             float* __restrict__ sums,
                             float* __restrict__ counts, int n) {
  int t = blockIdx.x * blockDim.x + threadIdx.x;
  int node = t >> 5;
  if (node >= n) return;
  int f = (t & 31) << 2;
  int g = batch[node];
  const float4 v = *(const float4*)(h + (size_t)node * DF + f);
  float* p = sums + (size_t)g * DF + f;
  atomicAdd(p + 0, v.x);
  atomicAdd(p + 1, v.y);
  atomicAdd(p + 2, v.z);
  atomicAdd(p + 3, v.w);
  if (f == 0) atomicAdd(counts + g, 1.0f);
}

// ------------------------------------------------------------- final linear D->1
__global__ void final_linear(const float* __restrict__ sums,
                             const float* __restrict__ counts,
                             const float* __restrict__ Wout,
                             const float* __restrict__ bout,
                             float* __restrict__ out) {
  __shared__ float s[DF];
  int g = blockIdx.x;
  int i = threadIdx.x;   // 0..127
  float cnt = counts[g];
  cnt = cnt > 1.0f ? cnt : 1.0f;
  s[i] = (sums[(size_t)g * DF + i] / cnt) * Wout[i];
  __syncthreads();
  for (int st = 64; st > 0; st >>= 1) {
    if (i < st) s[i] += s[i + st];
    __syncthreads();
  }
  if (i == 0) out[g] = s[0] + bout[0];
}

// =========================================================================
extern "C" void kernel_launch(void* const* d_in, const int* in_sizes, int n_in,
                              void* d_out, int out_size, void* d_ws, size_t ws_size,
                              hipStream_t stream) {
  const float* x     = (const float*)d_in[0];
  const int*   ei    = (const int*)d_in[1];
  const int*   batch = (const int*)d_in[2];
  const float* Wroot[4] = {(const float*)d_in[3], (const float*)d_in[6],
                           (const float*)d_in[9], (const float*)d_in[12]};
  const float* Wrel[4]  = {(const float*)d_in[4], (const float*)d_in[7],
                           (const float*)d_in[10], (const float*)d_in[13]};
  const float* bias[4]  = {(const float*)d_in[5], (const float*)d_in[8],
                           (const float*)d_in[11], (const float*)d_in[14]};
  const float* Wout = (const float*)d_in[15];
  const float* bout = (const float*)d_in[16];

  const int N = in_sizes[0] / DF;     // 40000
  const int E = in_sizes[1] / 2;      // 640000
  const int G = out_size;             // 128
  const int* src = ei;
  const int* dst = ei + E;

  // workspace carving (all regions 256B aligned)
  char* ws = (char*)d_ws;
  const size_t szF32 = (size_t)N * DF * sizeof(float);          // 20.48 MB
  const size_t szBF  = (size_t)N * DF * sizeof(unsigned short); // 10.24 MB
  float*          h     = (float*)(ws);
  float*          agg   = (float*)(ws + szF32);
  unsigned short* hb    = (unsigned short*)(ws + 2 * szF32);
  unsigned short* aggb  = (unsigned short*)(ws + 2 * szF32 + szBF);
  unsigned short* wb    = (unsigned short*)(ws + 2 * szF32 + 2 * szBF);   // 2*128*128 bf16
  float*          sums  = (float*)(ws + 2 * szF32 + 2 * szBF + 2 * DF * DF * sizeof(unsigned short));
  float*          cnts  = sums + (size_t)G * DF;

  const long nElem  = (long)N * DF;
  const long n4     = nElem / 4;
  const int  zBlk   = (int)((n4 + 255) / 256);
  const int  eBlk   = (E * 32 + 255) / 256;
  const int  cBlk   = (int)((n4 + 255) / 256);
  const int  wBlk   = (DF * DF / 4 + 255) / 256;
  const int  gBlk   = (N + 15) / 16;
  const int  pBlk   = (N * 32 + 255) / 256;

  for (int l = 0; l < 4; ++l) {
    const float* hin = (l == 0) ? x : h;
    // agg = segment_sum(hin[src] -> dst)
    zero_f32<<<zBlk, 256, 0, stream>>>(agg, nElem);
    edge_scatter<<<eBlk, 256, 0, stream>>>(hin, src, dst, agg, E);
    // bf16 copies for WMMA
    cvt_f32_bf16<<<cBlk, 256, 0, stream>>>(hin, (unsigned int*)hb, n4);
    cvt_f32_bf16<<<cBlk, 256, 0, stream>>>(agg, (unsigned int*)aggb, n4);
    cvt_f32_bf16<<<wBlk, 256, 0, stream>>>(Wroot[l], (unsigned int*)wb, DF * DF / 4);
    cvt_f32_bf16<<<wBlk, 256, 0, stream>>>(Wrel[l], (unsigned int*)(wb + DF * DF), DF * DF / 4);
    // h = gelu(hin@Wroot^T + agg@Wrel^T + b)   (writes f32 h; reads only bf16 copies)
    wmma_graphconv<<<gBlk, 256, 0, stream>>>(hb, aggb, wb, bias[l], h, N);
  }

  // global mean pool + output linear
  zero_f32<<<(int)(((long)G * DF + G + 255) / 256), 256, 0, stream>>>(sums, (long)G * DF + G);
  pool_scatter<<<pBlk, 256, 0, stream>>>(h, batch, sums, cnts, N);
  final_linear<<<G, DF, 0, stream>>>(sums, cnts, Wout, bout, (float*)d_out);
}
